// MemoryTransformerXL_42898133353133
// MI455X (gfx1250) — matrix operand order repaired and checked by reading, not compiled, not run on hardware
//
#include <hip/hip_runtime.h>
#include <hip/hip_bf16.h>

// ---------------- model constants ----------------
#define B_      2
#define DEPTH_  4
#define HEADS_  8
#define DIM_    1024
#define SEQ_    1024
#define MEMN_   1024
#define LMEM_   256
#define VOCAB_  32000
#define DH_     128
#define KV_     2304   // LMEM + MEM + SEQ
#define WINDOW_ 1280   // MEM + LMEM (causal mask offset)

typedef __attribute__((ext_vector_type(16))) __bf16 v16bf;
typedef __attribute__((ext_vector_type(4)))  __bf16 v4bf;
typedef __attribute__((ext_vector_type(8)))  float  v8f;

// ---------------- generic batched bf16 WMMA GEMM ----------------
// C = acc*alpha (+bias[n]) (+res)  or, in shift mode, C[m, n - shiftBase + m] += acc
// REQUIREMENT: M % 128 == 0, N % 128 == 0, K % 32 == 0, all lds/offsets % 4 == 0
// (holds for every launch in this model).
#define BM 128
#define BN 128
#define BK 32
#define LDK_S (BK + 8)   // padded K-stride (halfs); row pitch 80B (16B aligned)

__global__ __launch_bounds__(256)
void gemm_bf16_kernel(const float* __restrict__ A, const float* __restrict__ Bm,
                      float* C,
                      const float* __restrict__ bias, const float* res,
                      const float* __restrict__ alphaPtr,
                      int M, int N, int K, int lda, int ldb, int ldc,
                      int transA, int transB,
                      long long aO, long long aI, int aC,
                      long long bO, long long bI, int bC,
                      long long cO, long long cI, int cC,
                      int shiftMode, int shiftBase)
{
    __shared__ __bf16 As[BM][LDK_S];   // [m][k]
    __shared__ __bf16 Bs[BN][LDK_S];   // [n][k]  (transposed tile -> contiguous frags)

    const int z = blockIdx.z;
    const float* Ab = A  + (long long)(z / aC) * aO + (long long)(z % aC) * aI;
    const float* Bb = Bm + (long long)(z / bC) * bO + (long long)(z % bC) * bI;
    const long long cOff = (long long)(z / cC) * cO + (long long)(z % cC) * cI;
    float* Cb = C + cOff;
    const float* Rb = res ? (res + cOff) : (const float*)nullptr;

    const int tid  = threadIdx.x;
    const int lane = tid & 31;
    const int wave = tid >> 5;
    const int lr   = lane & 15;   // row (A) / col (B,C) within 16
    const int lh   = lane >> 4;   // half-wave select
    const int wRow = (wave >> 1) * 32;
    const int wCol = (wave & 1) * 64;
    const int by = blockIdx.y * BM;
    const int bx = blockIdx.x * BN;

    // per-thread load coordinates (float4 granularity)
    const int rowQ = tid >> 3;           // 0..31, step 32 per pass (along-K loads)
    const int kQ   = (tid & 7) * 4;      // 0..28
    const int kS   = tid >> 5;           // 0..7, step 8 per pass (along-MN loads)
    const int mnS  = (tid & 31) * 4;     // 0..124

    // ---- hoist global load offsets: uniform base (SGPR) + per-thread i32 offset ----
    // (keeps addrspace inference intact -> global_load_b128 GVS form, no DScnt tick)
    int offA[4], offB[4];
    int stepA, stepB;
    if (!transA) {
#pragma unroll
        for (int i = 0; i < 4; ++i) offA[i] = (by + rowQ + i * 32) * lda + kQ;
        stepA = BK;
    } else {
#pragma unroll
        for (int i = 0; i < 4; ++i) offA[i] = (kS + i * 8) * lda + (by + mnS);
        stepA = BK * lda;
    }
    if (!transB) {
#pragma unroll
        for (int i = 0; i < 4; ++i) offB[i] = (kS + i * 8) * ldb + (bx + mnS);
        stepB = BK * ldb;
    } else {
#pragma unroll
        for (int i = 0; i < 4; ++i) offB[i] = (bx + rowQ + i * 32) * ldb + kQ;
        stepB = BK;
    }
    // ---- hoist LDS store targets (fixed per thread) ----
    __bf16* sAvec = &As[rowQ][kQ];           // !transA: vector stores, row step 32
    __bf16* sAscl = &As[mnS][kS];            //  transA: scalar stores, offsets d*LDK_S + i*8
    __bf16* sBvec = &Bs[rowQ][kQ];           //  transB: vector stores
    __bf16* sBscl = &Bs[mnS][kS];            // !transB: scalar stores

    v8f acc[2][4];
#pragma unroll
    for (int i = 0; i < 2; ++i)
#pragma unroll
        for (int j = 0; j < 4; ++j)
#pragma unroll
            for (int v = 0; v < 8; ++v) acc[i][j][v] = 0.0f;

    const int kTiles = K / BK;
    for (int t = 0; t < kTiles; ++t) {
        // ---- stage next tile into registers (overlaps prior tile's WMMAs) ----
        float4 ra[4], rb[4];
#pragma unroll
        for (int i = 0; i < 4; ++i) { ra[i] = *(const float4*)(Ab + offA[i]); offA[i] += stepA; }
#pragma unroll
        for (int i = 0; i < 4; ++i) { rb[i] = *(const float4*)(Bb + offB[i]); offB[i] += stepB; }
        __syncthreads();   // previous tile's fragment reads complete

        // ---- convert + store to LDS (fixed addresses) ----
        if (!transA) {
#pragma unroll
            for (int i = 0; i < 4; ++i) {
                v4bf c = { (__bf16)ra[i].x, (__bf16)ra[i].y, (__bf16)ra[i].z, (__bf16)ra[i].w };
                *(v4bf*)(sAvec + i * 32 * LDK_S) = c;
            }
        } else {
#pragma unroll
            for (int i = 0; i < 4; ++i) {
                sAscl[0 * LDK_S + i * 8] = (__bf16)ra[i].x;
                sAscl[1 * LDK_S + i * 8] = (__bf16)ra[i].y;
                sAscl[2 * LDK_S + i * 8] = (__bf16)ra[i].z;
                sAscl[3 * LDK_S + i * 8] = (__bf16)ra[i].w;
            }
        }
        if (!transB) {
#pragma unroll
            for (int i = 0; i < 4; ++i) {
                sBscl[0 * LDK_S + i * 8] = (__bf16)rb[i].x;
                sBscl[1 * LDK_S + i * 8] = (__bf16)rb[i].y;
                sBscl[2 * LDK_S + i * 8] = (__bf16)rb[i].z;
                sBscl[3 * LDK_S + i * 8] = (__bf16)rb[i].w;
            }
        } else {
#pragma unroll
            for (int i = 0; i < 4; ++i) {
                v4bf c = { (__bf16)rb[i].x, (__bf16)rb[i].y, (__bf16)rb[i].z, (__bf16)rb[i].w };
                *(v4bf*)(sBvec + i * 32 * LDK_S) = c;
            }
        }
        __syncthreads();

        // ---- fragments (contiguous in LDS -> ds_load_b128) + WMMA ----
        // A frag (16x32): row = lane&15; K = e + 16*lh -> contiguous 16 halfs (matches B)
        v16bf af[2];
#pragma unroll
        for (int i = 0; i < 2; ++i) {
            int row = wRow + i * 16 + lr;
#pragma unroll
            for (int e = 0; e < 16; ++e) {
                int kk = e + 16 * lh;
                af[i][e] = As[row][kk];
            }
        }
        // B frag (32x16): col = lane&15; K = e + 16*lh -> contiguous 16 halfs
        v16bf bfv[4];
#pragma unroll
        for (int j = 0; j < 4; ++j) {
            int col = wCol + j * 16 + lr;
#pragma unroll
            for (int e = 0; e < 16; ++e)
                bfv[j][e] = Bs[col][e + 16 * lh];
        }
#pragma unroll
        for (int i = 0; i < 2; ++i)
#pragma unroll
            for (int j = 0; j < 4; ++j)
                acc[i][j] = __builtin_amdgcn_wmma_f32_16x16x32_bf16(
                    false, af[i], false, bfv[j], (short)0, acc[i][j], false, false);

        __syncthreads();
    }

    const float alpha = alphaPtr ? alphaPtr[0] : 1.0f;
#pragma unroll
    for (int i = 0; i < 2; ++i)
#pragma unroll
        for (int j = 0; j < 4; ++j)
#pragma unroll
            for (int v = 0; v < 8; ++v) {
                int gm = by + wRow + i * 16 + v + 8 * lh;   // C: m = v + 8*half
                int gn = bx + wCol + j * 16 + lr;           // C: n = lane&15
                float val = acc[i][j][v];
                if (shiftMode) {
                    int ct = gn - shiftBase + gm;           // rel_shift scatter-add
                    if (ct >= 0 && ct < N)
                        Cb[gm * ldc + ct] += val;
                } else {
                    float o = val * alpha;
                    if (bias) o += bias[gn];
                    if (Rb)   o += Rb[gm * ldc + gn];
                    Cb[gm * ldc + gn] = o;
                }
            }
}

// ---------------- elementwise / reduction kernels ----------------
__global__ void embed_kernel(float* h, const int* x, const float* emb, long long n4) {
    long long i = (long long)blockIdx.x * blockDim.x + threadIdx.x;
    long long stride = (long long)gridDim.x * blockDim.x;
    for (; i < n4; i += stride) {
        long long t = i >> 8;                 // DIM/4 = 256 float4 per token
        int d = (int)(i & 255) * 4;
        *(float4*)(h + i * 4) = *(const float4*)(emb + (long long)x[t] * DIM_ + d);
    }
}

__global__ void copy4_kernel(float* dst, const float* src, long long n4) {
    long long i = (long long)blockIdx.x * blockDim.x + threadIdx.x;
    long long stride = (long long)gridDim.x * blockDim.x;
    for (; i < n4; i += stride)
        *(float4*)(dst + i * 4) = *(const float4*)(src + i * 4);
}

__global__ void gelu_kernel(float* p, long long n) {
    long long i = (long long)blockIdx.x * blockDim.x + threadIdx.x;
    long long stride = (long long)gridDim.x * blockDim.x;
    for (; i < n; i += stride) {
        float v = p[i];
        p[i] = 0.5f * v * (1.0f + erff(v * 0.70710678118654752f));
    }
}

__global__ __launch_bounds__(256)
void layernorm_kernel(const float* x, float* y, const float* g, const float* b, int D) {
    const float* xr = x + (long long)blockIdx.x * D;
    float* yr = y + (long long)blockIdx.x * D;
    __shared__ float red[256];
    int t = threadIdx.x;
    float s = 0.0f;
    for (int i = t; i < D; i += 256) s += xr[i];
    red[t] = s; __syncthreads();
    for (int w = 128; w > 0; w >>= 1) { if (t < w) red[t] += red[t + w]; __syncthreads(); }
    float mu = red[0] / (float)D; __syncthreads();
    float vs = 0.0f;
    for (int i = t; i < D; i += 256) { float d = xr[i] - mu; vs += d * d; }
    red[t] = vs; __syncthreads();
    for (int w = 128; w > 0; w >>= 1) { if (t < w) red[t] += red[t + w]; __syncthreads(); }
    float inv = rsqrtf(red[0] / (float)D + 1e-5f);
    for (int i = t; i < D; i += 256) {
        float v = (xr[i] - mu) * inv;
        if (g) v = v * g[i] + b[i];
        yr[i] = v;
    }
}

// softmax over a dots row; dots already holds qk + shifted q*pe (unscaled)
__global__ __launch_bounds__(256)
void attn_softmax_kernel(float* dots, int seq, int kv, int window, float scale) {
    long long row = blockIdx.x;
    int r = (int)(row % seq);
    float* p = dots + row * (long long)kv;
    __shared__ float vals[KV_];
    __shared__ float red[256];
    int t = threadIdx.x;
    float mx = -3.402823466e38f;
    for (int c = t; c < kv; c += 256) {
        float v = (c - r <= window) ? scale * p[c] : -3.402823466e38f;
        vals[c] = v;
        mx = fmaxf(mx, v);
    }
    red[t] = mx; __syncthreads();
    for (int w = 128; w > 0; w >>= 1) { if (t < w) red[t] = fmaxf(red[t], red[t + w]); __syncthreads(); }
    mx = red[0]; __syncthreads();
    float sum = 0.0f;
    for (int c = t; c < kv; c += 256) { float e = __expf(vals[c] - mx); vals[c] = e; sum += e; }
    red[t] = sum; __syncthreads();
    for (int w = 128; w > 0; w >>= 1) { if (t < w) red[t] += red[t + w]; __syncthreads(); }
    float inv = 1.0f / red[0];
    for (int c = t; c < kv; c += 256) p[c] = vals[c] * inv;
}

// softmax over a contiguous 128-wide head segment (memory-net q)
__global__ __launch_bounds__(128)
void head_softmax_kernel(float* q, float scale) {
    float* p = q + (long long)blockIdx.x * 128;
    __shared__ float red[128];
    int t = threadIdx.x;
    float v = p[t] * scale;
    red[t] = v; __syncthreads();
    for (int w = 64; w > 0; w >>= 1) { if (t < w) red[t] = fmaxf(red[t], red[t + w]); __syncthreads(); }
    float mx = red[0]; __syncthreads();
    float e = __expf(v - mx);
    red[t] = e; __syncthreads();
    for (int w = 64; w > 0; w >>= 1) { if (t < w) red[t] += red[t + w]; __syncthreads(); }
    p[t] = e / red[0];
}

// column softmax (memory-net k): softmax over `rows` with leading dim `ld`
__global__ __launch_bounds__(256)
void col_softmax_kernel(float* buf, int rows, int ld, int cols, float scale) {
    int mb = blockIdx.x / cols;
    int c  = blockIdx.x % cols;
    float* p = buf + (long long)mb * rows * ld + c;
    __shared__ float vals[KV_];
    __shared__ float red[256];
    int t = threadIdx.x;
    float mx = -3.402823466e38f;
    for (int n = t; n < rows; n += 256) {
        float v = p[(long long)n * ld] * scale;
        vals[n] = v; mx = fmaxf(mx, v);
    }
    red[t] = mx; __syncthreads();
    for (int w = 128; w > 0; w >>= 1) { if (t < w) red[t] = fmaxf(red[t], red[t + w]); __syncthreads(); }
    mx = red[0]; __syncthreads();
    float sum = 0.0f;
    for (int n = t; n < rows; n += 256) { float e = __expf(vals[n] - mx); vals[n] = e; sum += e; }
    red[t] = sum; __syncthreads();
    for (int w = 128; w > 0; w >>= 1) { if (t < w) red[t] += red[t + w]; __syncthreads(); }
    float inv = 1.0f / red[0];
    for (int n = t; n < rows; n += 256) p[(long long)n * ld] = vals[n] * inv;
}

// ---------------- host orchestration ----------------
static void launch_gemm(hipStream_t st,
    const float* A, const float* Bp, float* C,
    const float* bias, const float* res, const float* alphaPtr,
    int M, int N, int K, int lda, int ldb, int ldc,
    int tA, int tB, int nbatch,
    long long aO, long long aI, int aC,
    long long bO, long long bI, int bC,
    long long cO, long long cI, int cC,
    int shiftMode)
{
    dim3 grid((N + BN - 1) / BN, (M + BM - 1) / BM, nbatch);
    gemm_bf16_kernel<<<grid, 256, 0, st>>>(A, Bp, C, bias, res, alphaPtr,
        M, N, K, lda, ldb, ldc, tA, tB,
        aO, aI, aC, bO, bI, bC, cO, cI, cC, shiftMode, SEQ_ - 1);
}

static void launch_copy(hipStream_t st, float* dst, const float* src, long long n) {
    long long n4 = n >> 2;   // all copies are multiples of 4
    int blocks = (int)((n4 + 255) / 256);
    if (blocks > 4096) blocks = 4096;
    copy4_kernel<<<blocks, 256, 0, st>>>(dst, src, n4);
}

extern "C" void kernel_launch(void* const* d_in, const int* in_sizes, int n_in,
                              void* d_out, int out_size, void* d_ws, size_t ws_size,
                              hipStream_t stream)
{
    (void)in_sizes; (void)n_in; (void)out_size;
    const int*   x         = (const int*)d_in[0];
    const float* mem       = (const float*)d_in[1];
    const float* lmem      = (const float*)d_in[2];
    const float* token_emb = (const float*)d_in[3];
    const float* pos_emb   = (const float*)d_in[4];
    const float* attn_ln_g = (const float*)d_in[5];
    const float* attn_ln_b = (const float*)d_in[6];
    const float* attn_wq   = (const float*)d_in[7];
    const float* attn_wkv  = (const float*)d_in[8];
    const float* attn_wo   = (const float*)d_in[9];
    const float* attn_bo   = (const float*)d_in[10];
    const float* ff_ln_g   = (const float*)d_in[11];
    const float* ff_ln_b   = (const float*)d_in[12];
    const float* ff_w1     = (const float*)d_in[13];
    const float* ff_b1     = (const float*)d_in[14];
    const float* ff_w2     = (const float*)d_in[15];
    const float* ff_b2     = (const float*)d_in[16];
    const float* logits_w  = (const float*)d_in[17];
    const float* logits_b  = (const float*)d_in[18];
    const float* mem_wq    = (const float*)d_in[19];
    const float* mem_wkv   = (const float*)d_in[20];
    const float* mem_wo    = (const float*)d_in[21];
    const float* rezero    = (const float*)d_in[22];

    // workspace layout (floats), with phase-disjoint aliasing
    const long long H_OFF    = 0;                          // 2,097,152
    const long long HID_OFF  = 2097152;                    // 8,388,608
    const long long NORM_OFF = 10485760;                   // 2,097,152
    const long long Q_OFF    = 12582912;                   // 2,097,152
    const long long KVIN_OFF = 14680064;                   // 4,718,592
    const long long KVB_OFF  = 19398656;                   // 9,437,184
    const long long DOTS_OFF = 28835840;                   // 37,748,736 (also FF, kvm_in)
    const long long AO_OFF   = 66584576;                   // 2,097,152 (layer phase only)
    const long long KVM_OFF  = DOTS_OFF + 18874368;        // 37,748,736 (memory phase only)
    const long long WS_FLOATS = 85458944;
    if (ws_size < (size_t)WS_FLOATS * sizeof(float)) return;

    float* ws  = (float*)d_ws;
    float* out = (float*)d_out;
    float* h    = ws + H_OFF;
    float* hid  = ws + HID_OFF;
    float* nrm  = ws + NORM_OFF;
    float* q    = ws + Q_OFF;
    float* kvin = ws + KVIN_OFF;
    float* kvb  = ws + KVB_OFF;
    float* dots = ws + DOTS_OFF;
    float* ao   = ws + AO_OFF;
    float* ffb  = ws + DOTS_OFF;      // alias (dots free after attn@v)
    float* nl    = nrm;               // memory phase aliases
    float* qm    = q;
    float* kvmin = ws + DOTS_OFF;
    float* kvm   = ws + KVM_OFF;
    float* ctx   = ws + KVIN_OFF;
    float* outm  = ws + KVB_OFF;

    const float scale_attn = 0.08838834764831845f;  // DH^-0.5
    const float scale_mem  = 0.29730177875068026f;  // DH^-0.25
    const long long TOK = (long long)B_ * SEQ_ * DIM_;
    const long long LOGN = (long long)B_ * SEQ_ * VOCAB_;
    const long long MEMOUTN = (long long)DEPTH_ * B_ * MEMN_ * DIM_;

    embed_kernel<<<2048, 256, 0, stream>>>(h, x, token_emb, TOK >> 2);

    for (int i = 0; i < DEPTH_; ++i) {
        launch_copy(stream, hid + (long long)i * TOK, h, TOK);
        layernorm_kernel<<<B_ * SEQ_, 256, 0, stream>>>(h, nrm, attn_ln_g + i * DIM_, attn_ln_b + i * DIM_, DIM_);
        for (int b = 0; b < B_; ++b) {   // kv_in = [lmem | mem | a_in]
            float* dst = kvin + (long long)b * KV_ * DIM_;
            launch_copy(stream, dst, lmem + ((long long)i * B_ + b) * LMEM_ * DIM_, (long long)LMEM_ * DIM_);
            launch_copy(stream, dst + LMEM_ * DIM_, mem + ((long long)i * B_ + b) * MEMN_ * DIM_, (long long)MEMN_ * DIM_);
            launch_copy(stream, dst + (LMEM_ + MEMN_) * DIM_, nrm + (long long)b * SEQ_ * DIM_, (long long)SEQ_ * DIM_);
        }
        launch_gemm(stream, nrm, attn_wq + (long long)i * DIM_ * DIM_, q, nullptr, nullptr, nullptr,
                    B_ * SEQ_, DIM_, DIM_, DIM_, DIM_, DIM_, 0, 0, 1, 0,0,1, 0,0,1, 0,0,1, 0);
        launch_gemm(stream, kvin, attn_wkv + (long long)i * DIM_ * 2 * DIM_, kvb, nullptr, nullptr, nullptr,
                    B_ * KV_, 2 * DIM_, DIM_, DIM_, 2 * DIM_, 2 * DIM_, 0, 0, 1, 0,0,1, 0,0,1, 0,0,1, 0);
        // dots = q @ k^T per (b,h)
        launch_gemm(stream, q, kvb, dots, nullptr, nullptr, nullptr,
                    SEQ_, KV_, DH_, DIM_, 2 * DIM_, KV_, 0, 1, B_ * HEADS_,
                    (long long)SEQ_ * DIM_, DH_, HEADS_,
                    (long long)KV_ * 2 * DIM_, DH_, HEADS_,
                    (long long)SEQ_ * KV_, 0, 1, 0);
        // dots += rel_shift(q @ pe^T)  (scatter-add epilogue)
        launch_gemm(stream, q, pos_emb, dots, nullptr, nullptr, nullptr,
                    SEQ_, KV_, DH_, DIM_, DH_, KV_, 0, 1, B_ * HEADS_,
                    (long long)SEQ_ * DIM_, DH_, HEADS_,
                    0, (long long)KV_ * DH_, HEADS_,
                    (long long)SEQ_ * KV_, 0, 1, 1);
        attn_softmax_kernel<<<B_ * HEADS_ * SEQ_, 256, 0, stream>>>(dots, SEQ_, KV_, WINDOW_, scale_attn);
        // out = attn @ v per (b,h) -> (b, t, h*DH)
        launch_gemm(stream, dots, kvb + DIM_, ao, nullptr, nullptr, nullptr,
                    SEQ_, DH_, KV_, KV_, 2 * DIM_, DIM_, 0, 0, B_ * HEADS_,
                    (long long)SEQ_ * KV_, 0, 1,
                    (long long)KV_ * 2 * DIM_, DH_, HEADS_,
                    (long long)SEQ_ * DIM_, DH_, HEADS_, 0);
        launch_gemm(stream, ao, attn_wo + (long long)i * DIM_ * DIM_, h, attn_bo + i * DIM_, h, nullptr,
                    B_ * SEQ_, DIM_, DIM_, DIM_, DIM_, DIM_, 0, 0, 1, 0,0,1, 0,0,1, 0,0,1, 0);
        layernorm_kernel<<<B_ * SEQ_, 256, 0, stream>>>(h, nrm, ff_ln_g + i * DIM_, ff_ln_b + i * DIM_, DIM_);
        launch_gemm(stream, nrm, ff_w1 + (long long)i * DIM_ * 4 * DIM_, ffb, ff_b1 + (long long)i * 4 * DIM_, nullptr, nullptr,
                    B_ * SEQ_, 4 * DIM_, DIM_, DIM_, 4 * DIM_, 4 * DIM_, 0, 0, 1, 0,0,1, 0,0,1, 0,0,1, 0);
        gelu_kernel<<<8192, 256, 0, stream>>>(ffb, (long long)B_ * SEQ_ * 4 * DIM_);
        launch_gemm(stream, ffb, ff_w2 + (long long)i * 4 * DIM_ * DIM_, h, ff_b2 + i * DIM_, h, nullptr,
                    B_ * SEQ_, DIM_, 4 * DIM_, 4 * DIM_, DIM_, DIM_, 0, 0, 1, 0,0,1, 0,0,1, 0,0,1, 0);
    }

    // logits = h @ logits_w + logits_b
    launch_gemm(stream, h, logits_w, out, logits_b, nullptr, nullptr,
                B_ * SEQ_, VOCAB_, DIM_, DIM_, VOCAB_, VOCAB_, 0, 0, 1, 0,0,1, 0,0,1, 0,0,1, 0);

    // next_mem = hiddens (MEM == SEQ, so the concat tail is exactly the hiddens stack)
    launch_copy(stream, out + LOGN, hid, MEMOUTN);

    // ---------------- memory network ----------------
    layernorm_kernel<<<DEPTH_ * B_ * LMEM_, 256, 0, stream>>>(lmem, nl, nullptr, nullptr, DIM_);
    launch_gemm(stream, nl, mem_wq, qm, nullptr, nullptr, nullptr,
                DEPTH_ * B_ * LMEM_, DIM_, DIM_, DIM_, DIM_, DIM_, 0, 0, 1, 0,0,1, 0,0,1, 0,0,1, 0);
    head_softmax_kernel<<<DEPTH_ * B_ * LMEM_ * HEADS_, 128, 0, stream>>>(qm, scale_mem);
    for (int m = 0; m < DEPTH_; ++m)
        for (int b = 0; b < B_; ++b) {   // kv_in = [nl | mem | hiddens]
            long long mb = (long long)m * B_ + b;
            float* dst = kvmin + mb * KV_ * DIM_;
            launch_copy(stream, dst, nl + mb * LMEM_ * DIM_, (long long)LMEM_ * DIM_);
            launch_copy(stream, dst + LMEM_ * DIM_, mem + mb * MEMN_ * DIM_, (long long)MEMN_ * DIM_);
            launch_copy(stream, dst + (LMEM_ + MEMN_) * DIM_, hid + mb * SEQ_ * DIM_, (long long)SEQ_ * DIM_);
        }
    launch_gemm(stream, kvmin, mem_wkv, kvm, nullptr, nullptr, nullptr,
                DEPTH_ * B_ * KV_, 2 * DIM_, DIM_, DIM_, 2 * DIM_, 2 * DIM_, 0, 0, 1, 0,0,1, 0,0,1, 0,0,1, 0);
    col_softmax_kernel<<<DEPTH_ * B_ * DIM_, 256, 0, stream>>>(kvm, KV_, 2 * DIM_, DIM_, scale_mem);
    // ctx[d,e] = sum_n k[n,d] v[n,e]  per (m,b,h)  (transA)
    launch_gemm(stream, kvm, kvm + DIM_, ctx, nullptr, nullptr, nullptr,
                DH_, DH_, KV_, 2 * DIM_, 2 * DIM_, DH_, 1, 0, DEPTH_ * B_ * HEADS_,
                (long long)KV_ * 2 * DIM_, DH_, HEADS_,
                (long long)KV_ * 2 * DIM_, DH_, HEADS_,
                (long long)DH_ * DH_, 0, 1, 0);
    // out[n,e] = q[n,:] @ ctx  per (m,b,h)
    launch_gemm(stream, qm, ctx, outm, nullptr, nullptr, nullptr,
                LMEM_, DH_, DH_, DIM_, DH_, DIM_, 0, 0, DEPTH_ * B_ * HEADS_,
                (long long)LMEM_ * DIM_, DH_, HEADS_,
                (long long)DH_ * DH_, 0, 1,
                (long long)LMEM_ * DIM_, DH_, HEADS_, 0);
    // next_lmem = (out @ wo) * rezero + lmem
    launch_gemm(stream, outm, mem_wo, out + LOGN + MEMOUTN, nullptr, lmem, rezero,
                DEPTH_ * B_ * LMEM_, DIM_, DIM_, DIM_, DIM_, DIM_, 0, 0, 1, 0,0,1, 0,0,1, 0,0,1, 0);
}
